// RGCN_graph_19971597926924
// MI455X (gfx1250) — matrix operand order repaired and checked by reading, not compiled, not run on hardware
//
#include <hip/hip_runtime.h>
#include <hip/hip_bf16.h>

// Problem constants (match reference)
#define N_NODES 40000
#define E_EDGES 1280000
#define F_IN    64
#define H_DIM   128
#define R_REL   20
#define C_CLS   32
#define G_GR    256

typedef __attribute__((ext_vector_type(16))) __bf16 v16bf;
typedef __attribute__((ext_vector_type(8)))  float  v8f;

__device__ __forceinline__ unsigned short f2bf(float x) {
    union { float f; unsigned int u; } c; c.f = x;
    unsigned int u = c.u;
    return (unsigned short)((u + 0x7FFFu + ((u >> 16) & 1u)) >> 16);  // RNE
}

// ---------------------------------------------------------------------------
// Edge-degree counts: cnt[dst*R + type] += 1   (shared by both layers)
// ---------------------------------------------------------------------------
__global__ void count_kernel(const int* __restrict__ dst, const int* __restrict__ et,
                             float* __restrict__ cnt, int E) {
    int e = blockIdx.x * blockDim.x + threadIdx.x;
    if (e >= E) return;
    atomicAdd(&cnt[(size_t)dst[e] * R_REL + et[e]], 1.0f);
}

// ---------------------------------------------------------------------------
// Scatter-add: acc[dst*R + type, :] += feat[src, :]   (one wave per edge)
// ---------------------------------------------------------------------------
__global__ void scatter_kernel(const float* __restrict__ feat,
                               const int* __restrict__ src, const int* __restrict__ dst,
                               const int* __restrict__ et, float* __restrict__ acc,
                               int E, int Fd) {
    int t = blockIdx.x * blockDim.x + threadIdx.x;
    int e = t >> 5, lane = t & 31;
    if (e >= E) return;
    const float* in = feat + (size_t)src[e] * Fd;
    float* out = acc + ((size_t)dst[e] * R_REL + et[e]) * Fd;
    for (int f = lane; f < Fd; f += 32) atomicAdd(out + f, in[f]);
}

// ---------------------------------------------------------------------------
// Normalize means and pack A' = [agg | feat] to bf16, width K = (R+1)*Fd
// acc is a one-shot 410MB stream -> non-temporal loads.
// ---------------------------------------------------------------------------
__global__ void pack_a_kernel(const float* __restrict__ acc, const float* __restrict__ cnt,
                              const float* __restrict__ feat, unsigned short* __restrict__ A,
                              int Fd) {
    size_t K = (size_t)(R_REL + 1) * Fd;
    size_t idx = (size_t)blockIdx.x * blockDim.x + threadIdx.x;
    if (idx >= (size_t)N_NODES * K) return;
    size_t n = idx / K, k = idx % K;
    float v;
    if (k < (size_t)R_REL * Fd) {
        int r = (int)(k / Fd);
        v = __builtin_nontemporal_load(&acc[(size_t)n * R_REL * Fd + k]) /
            fmaxf(cnt[(size_t)n * R_REL + r], 1.0f);
    } else {
        v = feat[(size_t)n * Fd + (k - (size_t)R_REL * Fd)];
    }
    A[idx] = f2bf(v);
}

// ---------------------------------------------------------------------------
// Pack W' = [W ; root] transposed to WT[H, K] in bf16 (B fragments contiguous)
// W is [R, Fd, H] row-major, root is [Fd, H]; both have trailing dim H.
// ---------------------------------------------------------------------------
__global__ void pack_w_kernel(const float* __restrict__ W, const float* __restrict__ root,
                              unsigned short* __restrict__ WT, int Fd) {
    int K = (R_REL + 1) * Fd;
    int idx = blockIdx.x * blockDim.x + threadIdx.x;
    if (idx >= H_DIM * K) return;
    int h = idx / K, k = idx % K;
    float v = (k < R_REL * Fd) ? W[(size_t)k * H_DIM + h]
                               : root[(size_t)(k - R_REL * Fd) * H_DIM + h];
    WT[(size_t)h * K + k] = f2bf(v);
}

// ---------------------------------------------------------------------------
// bf16 WMMA GEMM: out[M,128] = relu(A[M,K] @ WT^T + bias), fp32 accumulate.
// One wave -> 16 rows x 128 cols (8 wmma accumulators). M % 16 == 0,
// K % 64 == 0 (two-stage software pipeline, no remainder).
//
// A fragment (16x32 bf16): lane r=lane&15 -> row M=r, half=lane>>4,
//   VGPR i holds K-pair at (i>>2)*16 + (i&3)*2 + 8*half.
// B fragment (32x16 bf16): lane -> col N=r,
//   VGPR j holds K-pair at (j>>2)*16 + (j&3)*2 + 8*half.  WT[H,K] makes pairs contiguous.
// Double-buffered fragments (a0/b0, a1/b1) so loads for step k+32 overlap
// the 8 WMMAs of step k instead of s_wait_loadcnt 0 before every WMMA.
// ---------------------------------------------------------------------------
template <bool RELU>
__global__ void __launch_bounds__(128)
wmma_gemm_kernel(const unsigned short* __restrict__ A,
                 const unsigned short* __restrict__ WT,
                 const float* __restrict__ bias,
                 float* __restrict__ out, int M, int K) {
    const int wave = threadIdx.x >> 5;
    const int lane = threadIdx.x & 31;
    const int r = lane & 15;
    const int half = lane >> 4;
    const int m0 = (blockIdx.x * 4 + wave) * 16;
    if (m0 >= M) return;

    v8f acc[8];
#pragma unroll
    for (int nt = 0; nt < 8; ++nt) acc[nt] = (v8f){0, 0, 0, 0, 0, 0, 0, 0};

    // Per-lane base pointers with the half-offset folded in; remaining
    // per-VGPR offsets are compile-time immediates -> b128-coalesced loads.
    const unsigned short* Ar = A + (size_t)(m0 + r) * K + 8 * half;   // A stream: NT
    const unsigned short* Bp[8];
#pragma unroll
    for (int nt = 0; nt < 8; ++nt)
        Bp[nt] = WT + (size_t)(nt * 16 + r) * K + 8 * half;           // hot: cached

    union F { v16bf v; unsigned int u[8]; };
    F a0, a1, b0[8], b1[8];

    auto load_set = [&](F& a, F (&b)[8], int k) {
#pragma unroll
        for (int i = 0; i < 8; ++i)
            a.u[i] = __builtin_nontemporal_load(
                (const unsigned int*)(Ar + k + ((i >> 2) * 16 + (i & 3) * 2)));
#pragma unroll
        for (int nt = 0; nt < 8; ++nt)
#pragma unroll
            for (int i = 0; i < 8; ++i)
                b[nt].u[i] = *(const unsigned int*)(Bp[nt] + k + ((i >> 2) * 16 + (i & 3) * 2));
    };
    auto compute = [&](F& a, F (&b)[8]) {
#pragma unroll
        for (int nt = 0; nt < 8; ++nt)
            acc[nt] = __builtin_amdgcn_wmma_f32_16x16x32_bf16(
                false, a.v, false, b[nt].v, (short)0, acc[nt], false, false);
    };

    const int nsteps = K >> 5;          // even (K % 64 == 0)
    load_set(a0, b0, 0);
    for (int s = 0; s + 2 <= nsteps; s += 2) {
        load_set(a1, b1, (s + 1) << 5);
        compute(a0, b0);
        if (s + 2 < nsteps) load_set(a0, b0, (s + 2) << 5);
        compute(a1, b1);
    }

#pragma unroll
    for (int nt = 0; nt < 8; ++nt) {
        int col = nt * 16 + r;
        float bv = bias[col];
#pragma unroll
        for (int j = 0; j < 8; ++j) {
            int row = m0 + j + 8 * half;
            float v = acc[nt][j] + bv;
            if (RELU) v = fmaxf(v, 0.0f);
            out[(size_t)row * H_DIM + col] = v;
        }
    }
}

// ---------------------------------------------------------------------------
// Global mean pool: psum[batch[n], :] += h[n, :] ; pcnt[batch[n]] += 1
// ---------------------------------------------------------------------------
__global__ void pool_kernel(const float* __restrict__ h, const int* __restrict__ batch,
                            float* __restrict__ psum, float* __restrict__ pcnt) {
    int t = blockIdx.x * blockDim.x + threadIdx.x;
    int n = t >> 5, lane = t & 31;
    if (n >= N_NODES) return;
    int g = batch[n];
    const float* row = h + (size_t)n * H_DIM;
    for (int c = lane; c < H_DIM; c += 32) atomicAdd(&psum[(size_t)g * H_DIM + c], row[c]);
    if (lane == 0) atomicAdd(&pcnt[g], 1.0f);
}

// ---------------------------------------------------------------------------
// Final FC: out[g,c] = (psum[g,:]/max(pcnt,1)) @ fcw[:,c] + fcb[c]
// ---------------------------------------------------------------------------
__global__ void fc_kernel(const float* __restrict__ psum, const float* __restrict__ pcnt,
                          const float* __restrict__ fcw, const float* __restrict__ fcb,
                          float* __restrict__ out) {
    int t = blockIdx.x * blockDim.x + threadIdx.x;
    if (t >= G_GR * C_CLS) return;
    int g = t / C_CLS, c = t % C_CLS;
    float inv = 1.0f / fmaxf(pcnt[g], 1.0f);
    float s = 0.0f;
    const float* p = psum + (size_t)g * H_DIM;
#pragma unroll 8
    for (int k = 0; k < H_DIM; ++k) s += p[k] * fcw[(size_t)k * C_CLS + c];
    out[t] = s * inv + fcb[c];
}

// ---------------------------------------------------------------------------
extern "C" void kernel_launch(void* const* d_in, const int* in_sizes, int n_in,
                              void* d_out, int out_size, void* d_ws, size_t ws_size,
                              hipStream_t stream) {
    const float* x     = (const float*)d_in[0];
    const int*   ei    = (const int*)d_in[1];     // [2, E] flat: src then dst
    const int*   et    = (const int*)d_in[2];
    const int*   batch = (const int*)d_in[3];
    const float* W1    = (const float*)d_in[4];
    const float* root1 = (const float*)d_in[5];
    const float* b1    = (const float*)d_in[6];
    const float* W2    = (const float*)d_in[7];
    const float* root2 = (const float*)d_in[8];
    const float* b2    = (const float*)d_in[9];
    const float* fcw   = (const float*)d_in[10];
    const float* fcb   = (const float*)d_in[11];
    float* out = (float*)d_out;

    const int* src = ei;
    const int* dst = ei + E_EDGES;

    // Workspace carve-out (256B aligned)
    char* ws = (char*)d_ws;
    size_t off = 0;
    auto carve = [&](size_t bytes) -> void* {
        void* p = ws + off;
        off += (bytes + 255) & ~(size_t)255;
        return p;
    };
    float*          acc  = (float*)carve((size_t)N_NODES * R_REL * H_DIM * sizeof(float)); // 409.6MB (reused)
    float*          cnt  = (float*)carve((size_t)N_NODES * R_REL * sizeof(float));
    unsigned short* Abf  = (unsigned short*)carve((size_t)N_NODES * (R_REL + 1) * H_DIM * 2); // 215MB (reused)
    unsigned short* WT   = (unsigned short*)carve((size_t)(R_REL + 1) * H_DIM * H_DIM * 2);
    float*          h1   = (float*)carve((size_t)N_NODES * H_DIM * sizeof(float));
    float*          h2   = (float*)carve((size_t)N_NODES * H_DIM * sizeof(float));
    float*          psum = (float*)carve((size_t)G_GR * H_DIM * sizeof(float));
    float*          pcnt = (float*)carve((size_t)G_GR * sizeof(float));

    const int TB = 256;
    const size_t K1 = (size_t)(R_REL + 1) * F_IN;   // 1344
    const size_t K2 = (size_t)(R_REL + 1) * H_DIM;  // 2688
    const int gemm_blocks = (N_NODES / 16) / 4;     // 2500 tiles / 4 waves

    // Edge counts (shared by both layers)
    hipMemsetAsync(cnt, 0, (size_t)N_NODES * R_REL * sizeof(float), stream);
    count_kernel<<<(E_EDGES + TB - 1) / TB, TB, 0, stream>>>(dst, et, cnt, E_EDGES);

    // ---------------- Layer 1 ----------------
    hipMemsetAsync(acc, 0, (size_t)N_NODES * R_REL * F_IN * sizeof(float), stream);
    scatter_kernel<<<((size_t)E_EDGES * 32 + TB - 1) / TB, TB, 0, stream>>>(
        x, src, dst, et, acc, E_EDGES, F_IN);
    pack_w_kernel<<<(int)((H_DIM * K1 + TB - 1) / TB), TB, 0, stream>>>(W1, root1, WT, F_IN);
    pack_a_kernel<<<(int)(((size_t)N_NODES * K1 + TB - 1) / TB), TB, 0, stream>>>(
        acc, cnt, x, Abf, F_IN);
    wmma_gemm_kernel<true><<<gemm_blocks, 128, 0, stream>>>(Abf, WT, b1, h1, N_NODES, (int)K1);

    // ---------------- Layer 2 ----------------
    hipMemsetAsync(acc, 0, (size_t)N_NODES * R_REL * H_DIM * sizeof(float), stream);
    scatter_kernel<<<((size_t)E_EDGES * 32 + TB - 1) / TB, TB, 0, stream>>>(
        h1, src, dst, et, acc, E_EDGES, H_DIM);
    pack_w_kernel<<<(int)((H_DIM * K2 + TB - 1) / TB), TB, 0, stream>>>(W2, root2, WT, H_DIM);
    pack_a_kernel<<<(int)(((size_t)N_NODES * K2 + TB - 1) / TB), TB, 0, stream>>>(
        acc, cnt, h1, Abf, H_DIM);
    wmma_gemm_kernel<true><<<gemm_blocks, 128, 0, stream>>>(Abf, WT, b2, h2, N_NODES, (int)K2);

    // ---------------- Pool + FC ----------------
    hipMemsetAsync(psum, 0, (size_t)G_GR * H_DIM * sizeof(float), stream);
    hipMemsetAsync(pcnt, 0, (size_t)G_GR * sizeof(float), stream);
    pool_kernel<<<((size_t)N_NODES * 32 + TB - 1) / TB, TB, 0, stream>>>(h2, batch, psum, pcnt);
    fc_kernel<<<(G_GR * C_CLS + TB - 1) / TB, TB, 0, stream>>>(psum, pcnt, fcw, fcb, out);
}
